// GINModel_7035156431319
// MI455X (gfx1250) — matrix operand order repaired and checked by reading, not compiled, not run on hardware
//
#include <hip/hip_runtime.h>
#include <hip/hip_bf16.h>

// ---------------------------------------------------------------------------
// GIN forward for gfx1250: bf16 WMMA GEMMs (fragment-native packed operands)
// + f32 atomics for graph aggregation / pooling.
// ---------------------------------------------------------------------------

#define N_NODES  50000
#define N_EDGES  600000
#define NODE_DIM 32
#define HID      128
#define N_LAYERS 4
#define N_GRAPHS 256
#define BN_EPS   1e-5f

typedef __attribute__((ext_vector_type(16))) __bf16          v16bf;
typedef __attribute__((ext_vector_type(8)))  float           v8f;
typedef __attribute__((ext_vector_type(8)))  unsigned short  v8us;

union BF16x16 { unsigned short s[16]; v8us h[2]; v16bf v; };

__device__ __forceinline__ unsigned short f2bf(float f) {
    unsigned int u = __float_as_uint(f);
    unsigned int r = u + 0x7FFFu + ((u >> 16) & 1u);   // round-to-nearest-even
    return (unsigned short)(r >> 16);
}

// ---------------------------------------------------------------------------
// Utility kernels
// ---------------------------------------------------------------------------
__global__ void fill_zero(float* __restrict__ p, size_t n) {
    size_t i = (size_t)blockIdx.x * blockDim.x + threadIdx.x;
    size_t stride = (size_t)gridDim.x * blockDim.x;
    for (; i < n; i += stride) p[i] = 0.0f;
}

__global__ void f32_to_bf16_kernel(const float* __restrict__ in,
                                   unsigned short* __restrict__ out, int n) {
    int i = blockIdx.x * blockDim.x + threadIdx.x;
    if (i < n) out[i] = f2bf(in[i]);
}

// Pack f32 weights W[K,N] (row-major) into fragment-native bf16 layout:
//   out[((k/32)*N + n)*32 + (k%32)] = bf16(W[k*N + n])
// so a B fragment for (kstep, ncol) is 32 contiguous bf16 (two 16B loads).
__global__ void pack_b_bf16(const float* __restrict__ W,
                            unsigned short* __restrict__ out, int K, int N) {
    int i = blockIdx.x * blockDim.x + threadIdx.x;
    if (i >= K * N) return;
    int t    = i & 31;          // k within kstep
    int rest = i >> 5;
    int n    = rest % N;
    int ks   = rest / N;
    out[i] = f2bf(W[(size_t)(ks * 32 + t) * N + n]);
}

// ---------------------------------------------------------------------------
// Tiled GEMM: C[m,n] = sum_k A[m,k]*B[k,n] + bias[n]
// A: bf16 row-major [M,K]; Bp: fragment-packed bf16 (see pack_b_bf16).
// One wave computes a 16x64 strip: one A fragment reused by 4 WMMAs per kstep.
// K, N compile-time; M % 16 == 0, K % 32 == 0, N % 64 == 0.
// ---------------------------------------------------------------------------
template<int K, int N>
__global__ void gemm_bf16_wmma(const unsigned short* __restrict__ A,
                               const unsigned short* __restrict__ Bp,
                               const float* __restrict__ bias,
                               float* __restrict__ C, int M) {
    constexpr int tilesN = N / 64;
    const int wave = threadIdx.x >> 5;
    const int lane = threadIdx.x & 31;
    const int totalTiles = (M >> 4) * tilesN;
    const int tile = blockIdx.x * (blockDim.x >> 5) + wave;
    if (tile >= totalTiles) return;            // wave-uniform: EXEC stays all-ones

    const int tm   = tile / tilesN;
    const int tn   = tile - tm * tilesN;       // 64-column group
    const int half = lane >> 4;                // 0: lanes 0-15, 1: lanes 16-31
    const int l16  = lane & 15;

    const unsigned short* __restrict__ Arow = A + (size_t)((tm << 4) + l16) * K;

    v8f acc0 = {}, acc1 = {}, acc2 = {}, acc3 = {};

    #pragma unroll
    for (int k0 = 0; k0 < K; k0 += 32) {
        // A 16x32 bf16 fragment: elems 0-15 -> K=k0+half*16+i
        BF16x16 a;
        a.h[0] = *(const v8us*)(Arow + k0 + half * 16);
        a.h[1] = *(const v8us*)(Arow + k0 + half * 16 + 8);

        const int ks = k0 >> 5;
        #pragma unroll
        for (int j = 0; j < 4; ++j) {
            const int ncol = (tn << 6) + (j << 4) + l16;
            const unsigned short* bp =
                Bp + ((size_t)ks * N + ncol) * 32 + half * 16;
            BF16x16 b;
            b.h[0] = *(const v8us*)(bp);
            b.h[1] = *(const v8us*)(bp + 8);
            v8f& acc = (j == 0) ? acc0 : (j == 1) ? acc1 : (j == 2) ? acc2 : acc3;
            acc = __builtin_amdgcn_wmma_f32_16x16x32_bf16(
                      false, a.v, false, b.v, (short)0, acc, false, false);
        }
    }

    // D f32 16x16 layout: VGPR r -> M = r + 8*half, col = lane%16
    #pragma unroll
    for (int j = 0; j < 4; ++j) {
        const v8f& acc = (j == 0) ? acc0 : (j == 1) ? acc1 : (j == 2) ? acc2 : acc3;
        const int ncol = (tn << 6) + (j << 4) + l16;
        const float bn = bias ? bias[ncol] : 0.0f;
        #pragma unroll
        for (int r = 0; r < 8; ++r) {
            int row = (tm << 4) + r + half * 8;
            C[(size_t)row * N + ncol] = acc[r] + bn;
        }
    }
}

// ---------------------------------------------------------------------------
// Column statistics: coalesced row chunks + global atomics. blockDim.x == N.
// ---------------------------------------------------------------------------
__global__ void col_stats(const float* __restrict__ Y, int M, int N,
                          int rowsPerBlock,
                          float* __restrict__ sum, float* __restrict__ sumsq) {
    const int col = threadIdx.x;
    int r0 = blockIdx.x * rowsPerBlock;
    int r1 = r0 + rowsPerBlock; if (r1 > M) r1 = M;
    float s = 0.0f, q = 0.0f;
    for (int r = r0; r < r1; ++r) {
        float v = Y[(size_t)r * N + col];
        s += v;
        q += v * v;
    }
    atomicAdd(&sum[col], s);
    atomicAdd(&sumsq[col], q);
}

__global__ void bn_finalize(const float* __restrict__ sum,
                            const float* __restrict__ sumsq,
                            const float* __restrict__ g,
                            const float* __restrict__ beta,
                            int N, float invM,
                            float* __restrict__ scale, float* __restrict__ shift) {
    int n = threadIdx.x;
    if (n >= N) return;
    float m   = sum[n] * invM;
    float var = sumsq[n] * invM - m * m;
    if (var < 0.0f) var = 0.0f;
    float sc = g[n] * rsqrtf(var + BN_EPS);
    scale[n] = sc;
    shift[n] = beta[n] - m * sc;
}

// out[i] = (resid ? resid[i] : 0) + relu(Y[i]*scale[col] + shift[col]); may alias
__global__ void bn_relu_f32(const float* __restrict__ Y,
                            const float* __restrict__ scale,
                            const float* __restrict__ shift,
                            const float* __restrict__ resid,
                            float* __restrict__ out, size_t total, int N) {
    size_t i = (size_t)blockIdx.x * blockDim.x + threadIdx.x;
    if (i >= total) return;
    int col = (int)(i % (size_t)N);
    float v = fmaxf(Y[i] * scale[col] + shift[col], 0.0f);
    if (resid) v += resid[i];
    out[i] = v;
}

// bf16-output variant (feeds next GEMM's A operand directly)
__global__ void bn_relu_bf16(const float* __restrict__ Y,
                             const float* __restrict__ scale,
                             const float* __restrict__ shift,
                             unsigned short* __restrict__ out, size_t total, int N) {
    size_t i = (size_t)blockIdx.x * blockDim.x + threadIdx.x;
    if (i >= total) return;
    int col = (int)(i % (size_t)N);
    out[i] = f2bf(fmaxf(Y[i] * scale[col] + shift[col], 0.0f));
}

// zbf[i] = bf16((1+eps)*h[i] + agg[i])  (GEMM1 A operand)
__global__ void axpy_eps_bf16(const float* __restrict__ h,
                              const float* __restrict__ epsp,
                              const float* __restrict__ agg,
                              unsigned short* __restrict__ zbf, size_t total) {
    size_t i = (size_t)blockIdx.x * blockDim.x + threadIdx.x;
    if (i >= total) return;
    zbf[i] = f2bf((1.0f + epsp[0]) * h[i] + agg[i]);
}

// ---------------------------------------------------------------------------
// Edge aggregation: agg[dst[e]][:] += h[src[e]][:]; wave per edge, float4/lane.
// ---------------------------------------------------------------------------
__global__ void edge_agg(const int* __restrict__ src, const int* __restrict__ dst,
                         const float* __restrict__ h, float* __restrict__ agg,
                         int nEdges) {
    int tid  = blockIdx.x * blockDim.x + threadIdx.x;
    int e    = tid >> 5;
    int lane = tid & 31;
    if (e >= nEdges) return;
    int s = src[e], d = dst[e];
    float4 v = ((const float4*)(h + (size_t)s * HID))[lane];
    float* ad = agg + (size_t)d * HID + lane * 4;
    atomicAdd(ad + 0, v.x);
    atomicAdd(ad + 1, v.y);
    atomicAdd(ad + 2, v.z);
    atomicAdd(ad + 3, v.w);
}

// Graph pooling: s[batch[n]][:] += h[n][:], cnt[batch[n]] += 1
__global__ void pool_sum(const float* __restrict__ h, const int* __restrict__ batch,
                         float* __restrict__ s, float* __restrict__ cnt, int nNodes) {
    int tid  = blockIdx.x * blockDim.x + threadIdx.x;
    int node = tid >> 5;
    int lane = tid & 31;
    if (node >= nNodes) return;
    int b = batch[node];
    float4 v = ((const float4*)(h + (size_t)node * HID))[lane];
    float* sd = s + (size_t)b * HID + lane * 4;
    atomicAdd(sd + 0, v.x);
    atomicAdd(sd + 1, v.y);
    atomicAdd(sd + 2, v.z);
    atomicAdd(sd + 3, v.w);
    if (lane == 0) atomicAdd(cnt + b, 1.0f);
}

// pooled[g] = concat(s[g]/max(cnt,1), s[g])  -> [G, 2H]
__global__ void build_pooled(const float* __restrict__ s, const float* __restrict__ cnt,
                             float* __restrict__ pooled) {
    int g = blockIdx.x, c = threadIdx.x;            // c in [0,128)
    float sv = s[(size_t)g * HID + c];
    float cv = fmaxf(cnt[g], 1.0f);
    pooled[(size_t)g * (2 * HID) + c]       = sv / cv;
    pooled[(size_t)g * (2 * HID) + HID + c] = sv;
}

// ---------------------------------------------------------------------------
// Readout (tiny, ~17 MFLOP): plain fp32
// ---------------------------------------------------------------------------
__global__ void readout1(const float* __restrict__ pooled, const float* __restrict__ W1,
                         const float* __restrict__ b1, float* __restrict__ y1) {
    int n = threadIdx.x, g = blockIdx.x;
    float acc = b1[n];
    for (int k = 0; k < 2 * HID; ++k)
        acc += pooled[(size_t)g * (2 * HID) + k] * W1[(size_t)k * HID + n];
    y1[(size_t)g * HID + n] = acc;
}

__global__ void readout2(const float* __restrict__ y1, const float* __restrict__ W2,
                         const float* __restrict__ b2, float* __restrict__ y2) {
    int n = threadIdx.x, g = blockIdx.x;
    float acc = b2[n];
    for (int k = 0; k < HID; ++k)
        acc += y1[(size_t)g * HID + k] * W2[(size_t)k * (HID / 2) + n];
    y2[(size_t)g * (HID / 2) + n] = fmaxf(acc, 0.0f);
}

__global__ void readout3(const float* __restrict__ y2, const float* __restrict__ W3,
                         const float* __restrict__ b3, float* __restrict__ out) {
    int g = blockIdx.x * blockDim.x + threadIdx.x;
    if (g >= N_GRAPHS) return;
    float acc = b3[0];
    for (int k = 0; k < HID / 2; ++k)
        acc += y2[(size_t)g * (HID / 2) + k] * W3[k];
    out[g] = acc;
}

// ---------------------------------------------------------------------------
// Host orchestration
// ---------------------------------------------------------------------------
extern "C" void kernel_launch(void* const* d_in, const int* in_sizes, int n_in,
                              void* d_out, int out_size, void* d_ws, size_t ws_size,
                              hipStream_t stream) {
    (void)in_sizes; (void)n_in; (void)out_size; (void)ws_size;

    const float* x     = (const float*)d_in[0];
    const int*   ei    = (const int*)d_in[1];
    const int*   src   = ei;
    const int*   dst   = ei + N_EDGES;
    const int*   batch = (const int*)d_in[2];
    const float* projW    = (const float*)d_in[3];
    const float* projb    = (const float*)d_in[4];
    const float* projg    = (const float*)d_in[5];
    const float* projbeta = (const float*)d_in[6];
    // layer l base index = 7 + 9*l: eps,W1,b1,g1,beta1,W2,b2,g2,beta2
    const float* rW1    = (const float*)d_in[43];
    const float* rb1    = (const float*)d_in[44];
    const float* rg1    = (const float*)d_in[45];
    const float* rbeta1 = (const float*)d_in[46];
    const float* rW2    = (const float*)d_in[47];
    const float* rb2    = (const float*)d_in[48];
    const float* rW3    = (const float*)d_in[49];
    const float* rb3    = (const float*)d_in[50];

    // ---- workspace carve-up (256B aligned) ----
    size_t off = 0;
    char* wsb = (char*)d_ws;
    auto alloc = [&](size_t bytes) -> void* {
        void* p = (void*)(wsb + off);
        off += (bytes + 255) & ~(size_t)255;
        return p;
    };
    float* h    = (float*)alloc((size_t)N_NODES * HID * 4);           // 25.6 MB
    float* agg  = (float*)alloc((size_t)N_NODES * HID * 4);           // 25.6 MB (agg / y2)
    float* y1   = (float*)alloc((size_t)N_NODES * 2 * HID * 4);       // 51.2 MB
    unsigned short* xbf  = (unsigned short*)alloc((size_t)N_NODES * NODE_DIM * 2);   // 3.2 MB
    unsigned short* zbf  = (unsigned short*)alloc((size_t)N_NODES * HID * 2);        // 12.8 MB
    unsigned short* z1bf = (unsigned short*)alloc((size_t)N_NODES * 2 * HID * 2);    // 25.6 MB
    unsigned short* wprojp = (unsigned short*)alloc((size_t)NODE_DIM * HID * 2);
    unsigned short* w1p[N_LAYERS];
    unsigned short* w2p[N_LAYERS];
    for (int l = 0; l < N_LAYERS; ++l) {
        w1p[l] = (unsigned short*)alloc((size_t)HID * 2 * HID * 2);
        w2p[l] = (unsigned short*)alloc((size_t)2 * HID * HID * 2);
    }
    float* ssum  = (float*)alloc(512 * 4);    // ssum[256] | ssq[256] contiguous
    float* ssq   = ssum + 256;
    float* scale = (float*)alloc(256 * 4);
    float* shift = (float*)alloc(256 * 4);
    float* ps    = (float*)alloc(((size_t)N_GRAPHS * HID + N_GRAPHS) * 4); // s|cnt
    float* pcnt  = ps + (size_t)N_GRAPHS * HID;
    float* pooled = (float*)alloc((size_t)N_GRAPHS * 2 * HID * 4);
    float* ry1    = (float*)alloc((size_t)N_GRAPHS * HID * 4);
    float* ry2    = (float*)alloc((size_t)N_GRAPHS * (HID / 2) * 4);

    const size_t nodeH  = (size_t)N_NODES * HID;       // 6.4M
    const size_t node2H = (size_t)N_NODES * 2 * HID;   // 12.8M

    // ---- operand preparation ----
    f32_to_bf16_kernel<<<(N_NODES * NODE_DIM + 255) / 256, 256, 0, stream>>>(
        x, xbf, N_NODES * NODE_DIM);
    pack_b_bf16<<<(NODE_DIM * HID + 255) / 256, 256, 0, stream>>>(projW, wprojp, NODE_DIM, HID);
    for (int l = 0; l < N_LAYERS; ++l) {
        const float* W1 = (const float*)d_in[7 + 9 * l + 1];
        const float* W2 = (const float*)d_in[7 + 9 * l + 5];
        pack_b_bf16<<<(HID * 2 * HID + 255) / 256, 256, 0, stream>>>(W1, w1p[l], HID, 2 * HID);
        pack_b_bf16<<<(2 * HID * HID + 255) / 256, 256, 0, stream>>>(W2, w2p[l], 2 * HID, HID);
    }

    auto bn_stats = [&](const float* Y, int M, int N, const float* g, const float* beta) {
        fill_zero<<<2, 256, 0, stream>>>(ssum, 512);
        int blocks = (M + 255) / 256;
        col_stats<<<blocks, N, 0, stream>>>(Y, M, N, 256, ssum, ssq);
        bn_finalize<<<1, N, 0, stream>>>(ssum, ssq, g, beta, N, 1.0f / (float)M, scale, shift);
    };

    // ---- projection: h = relu(bn(x @ Wp + bp)) ----
    {
        int tiles = (N_NODES / 16) * (HID / 64);      // 6250 wave-tiles
        gemm_bf16_wmma<NODE_DIM, HID><<<(tiles + 7) / 8, 256, 0, stream>>>(
            xbf, wprojp, projb, agg, N_NODES);
        bn_stats(agg, N_NODES, HID, projg, projbeta);
        bn_relu_f32<<<(int)((nodeH + 255) / 256), 256, 0, stream>>>(
            agg, scale, shift, nullptr, h, nodeH, HID);
    }

    // ---- GIN layers ----
    for (int l = 0; l < N_LAYERS; ++l) {
        const float* epsp  = (const float*)d_in[7 + 9 * l + 0];
        const float* b1    = (const float*)d_in[7 + 9 * l + 2];
        const float* g1    = (const float*)d_in[7 + 9 * l + 3];
        const float* beta1 = (const float*)d_in[7 + 9 * l + 4];
        const float* b2    = (const float*)d_in[7 + 9 * l + 6];
        const float* g2    = (const float*)d_in[7 + 9 * l + 7];
        const float* beta2 = (const float*)d_in[7 + 9 * l + 8];

        fill_zero<<<8192, 256, 0, stream>>>(agg, nodeH);
        edge_agg<<<(N_EDGES * 32 + 255) / 256, 256, 0, stream>>>(src, dst, h, agg, N_EDGES);
        axpy_eps_bf16<<<(int)((nodeH + 255) / 256), 256, 0, stream>>>(h, epsp, agg, zbf, nodeH);

        // z @ W1 -> y1 [N, 2H]
        int tiles1 = (N_NODES / 16) * (2 * HID / 64); // 12500 wave-tiles
        gemm_bf16_wmma<HID, 2 * HID><<<(tiles1 + 7) / 8, 256, 0, stream>>>(
            zbf, w1p[l], b1, y1, N_NODES);
        bn_stats(y1, N_NODES, 2 * HID, g1, beta1);
        bn_relu_bf16<<<(int)((node2H + 255) / 256), 256, 0, stream>>>(
            y1, scale, shift, z1bf, node2H, 2 * HID);

        // z1 @ W2 -> agg [N, H]
        int tiles2 = (N_NODES / 16) * (HID / 64);     // 6250 wave-tiles
        gemm_bf16_wmma<2 * HID, HID><<<(tiles2 + 7) / 8, 256, 0, stream>>>(
            z1bf, w2p[l], b2, agg, N_NODES);
        bn_stats(agg, N_NODES, HID, g2, beta2);
        bn_relu_f32<<<(int)((nodeH + 255) / 256), 256, 0, stream>>>(
            agg, scale, shift, /*resid=*/h, h, nodeH, HID);   // h = h + relu(bn(.))
    }

    // ---- pooling ----
    fill_zero<<<256, 256, 0, stream>>>(ps, (size_t)N_GRAPHS * HID + N_GRAPHS);
    pool_sum<<<(N_NODES * 32 + 255) / 256, 256, 0, stream>>>(h, batch, ps, pcnt, N_NODES);
    build_pooled<<<N_GRAPHS, HID, 0, stream>>>(ps, pcnt, pooled);

    // ---- readout ----
    readout1<<<N_GRAPHS, HID, 0, stream>>>(pooled, rW1, rb1, ry1);
    bn_stats(ry1, N_GRAPHS, HID, rg1, rbeta1);
    bn_relu_f32<<<(N_GRAPHS * HID + 255) / 256, 256, 0, stream>>>(
        ry1, scale, shift, nullptr, ry1, (size_t)N_GRAPHS * HID, HID);
    readout2<<<N_GRAPHS, HID / 2, 0, stream>>>(ry1, rW2, rb2, ry2);
    readout3<<<1, N_GRAPHS, 0, stream>>>(ry2, rW3, rb3, (float*)d_out);
}
